// DotPredictor_14173392077066
// MI455X (gfx1250) — matrix operand order repaired and checked by reading, not compiled, not run on hardware
//
#include <hip/hip_runtime.h>
#include <hip/hip_bf16.h>

typedef __attribute__((ext_vector_type(16))) __bf16 bf16x16;
typedef __attribute__((ext_vector_type(8)))  __bf16 bf16x8;
typedef __attribute__((ext_vector_type(4)))  __bf16 bf16x4;
typedef __attribute__((ext_vector_type(8)))  float  v8f;
typedef __attribute__((ext_vector_type(4)))  unsigned int uint32x4;
typedef __attribute__((ext_vector_type(8)))  int  int32x8;
typedef __attribute__((ext_vector_type(4)))  int  int32x4;

#define ASTRIDE  264   // staging row stride in halves (132 dwords == 4 mod 64 -> conflict-free b128)
#define W1STRIDE 136   // W1 LDS row stride in halves (68 dwords == 4 mod 64)
#define W2STRIDE 264   // W2 LDS row stride in halves
// LDS layout: staging FIRST so all per-wave staging accesses fold into DS offset immediates.
#define ST_WAVE_BYTES (16 * ASTRIDE * 2)                  // 8,448
#define ST_LDS_BYTES  (8 * ST_WAVE_BYTES)                 // 67,584
#define W1_LDS_OFF    ST_LDS_BYTES                        // 67,584
#define W1_LDS_BYTES  (256 * W1STRIDE * 2)                // 69,632
#define W2_LDS_OFF    (W1_LDS_OFF + W1_LDS_BYTES)         // 137,216
#define W2_LDS_BYTES  (256 * W2STRIDE * 2)                // 135,168
#define LDS_TOTAL     (W2_LDS_OFF + W2_LDS_BYTES)         // 272,384

#if __has_builtin(__builtin_amdgcn_tensor_load_to_lds)
#define TDM_MODE 1
#else
#define TDM_MODE 0
#endif

// single-instruction ReLU: v_med3_f32 x, 0, +inf  (no canonicalize pair like fmax)
__device__ inline float relu1(float x) {
  return __builtin_amdgcn_fmed3f(x, 0.0f, __builtin_inff());
}

// ---------------------------------------------------------------------------
// Pre-pass: W1 [128][256] f32 -> W1t [256][128] bf16 ; W2 [256][256] -> W2t [256][256] bf16
// ---------------------------------------------------------------------------
__global__ __launch_bounds__(128) void prep_weights(const float* __restrict__ W1,
                                                    const float* __restrict__ W2,
                                                    unsigned short* __restrict__ w1t_u,
                                                    unsigned short* __restrict__ w2t_u) {
  __bf16* w1t = reinterpret_cast<__bf16*>(w1t_u);
  __bf16* w2t = reinterpret_cast<__bf16*>(w2t_u);
  int idx = blockIdx.x * blockDim.x + threadIdx.x;
  if (idx < 256 * 128) {                       // W1t[n][k] = W1[k][n]
    int n = idx >> 7, k = idx & 127;
    w1t[idx] = (__bf16)W1[k * 256 + n];
  }
  int idx2 = idx - 256 * 128;
  if (idx2 >= 0 && idx2 < 256 * 256) {         // W2t[n][k] = W2[k][n]
    int n = idx2 >> 8, k = idx2 & 255;
    w2t[idx2] = (__bf16)W2[k * 256 + n];
  }
}

// Build a 2-D TDM descriptor (group0) : count=1, user mode, type=2 ("image")
__device__ inline uint32x4 tdm_group0(unsigned lds_off, const void* gaddr) {
  const unsigned long long ga = (unsigned long long)(uintptr_t)gaddr;
  uint32x4 g0;
  g0[0] = 1u;                                             // count=1, no gather
  g0[1] = lds_off;                                        // lds_addr (bytes)
  g0[2] = (unsigned)(ga & 0xffffffffu);                   // global_addr[31:0]
  g0[3] = (unsigned)((ga >> 32) & 0x01ffffffu) | (2u << 30); // addr[56:32] | type=2
  return g0;
}
// group1: 2-D tensor of DWORDs (data_size=4B), whole-tensor tile, LDS row padding.
__device__ inline int32x8 tdm_group1(unsigned dim0, unsigned dim1,
                                     unsigned pad_interval_enc, unsigned pad_amount_enc) {
  int32x8 g1;
  g1[0] = (int)((2u << 16) | (1u << 20) |
                (pad_interval_enc << 22) | (pad_amount_enc << 25)); // data_size=4B, pad_enable
  g1[1] = (int)((dim0 & 0xffffu) << 16);                  // tensor_dim0[15:0] @ bits 63:48
  g1[2] = (int)((dim0 >> 16) | ((dim1 & 0xffffu) << 16)); // dim0[31:16] | dim1[15:0]
  g1[3] = (int)((dim1 >> 16) | ((dim0 & 0xffffu) << 16)); // dim1[31:16] | tile_dim0
  g1[4] = (int)(dim1 & 0xffffu);                          // tile_dim1 (tile_dim2 = 0)
  g1[5] = (int)dim0;                                      // tensor_dim0_stride[31:0] (contiguous)
  g1[6] = 0;                                              // stride0[47:32] | stride1[15:0]
  g1[7] = 0;
  return g1;
}

#if TDM_MODE
__device__ inline void tdm_load_2d(uint32x4 g0, int32x8 g1) {
  const int32x4 z4 = {0, 0, 0, 0};
  const int32x8 z8 = {0, 0, 0, 0, 0, 0, 0, 0};
  // clang-23 / therock signature: (v4u32 g0, v8i32 g1, v4i32 g2, v4i32 g3, v8i32, i32 cpol)
  __builtin_amdgcn_tensor_load_to_lds(g0, g1, z4, z4, z8, 0);
}
#endif

// ---------------------------------------------------------------------------
// Fused edge kernel. Block = 256 threads = 8 waves on one WGP; weights LDS-resident
// (staged by the Tensor Data Mover with D# row padding).  Each wave owns 2 row
// tiles (32 edges); layer-2 B fragments are reused across both tiles.
// ---------------------------------------------------------------------------
__global__ __launch_bounds__(256) void edge_mlp(
    const float* __restrict__ h, const int* __restrict__ src, const int* __restrict__ dst,
    const unsigned short* __restrict__ w1t_u, const unsigned short* __restrict__ w2t_u,
    const float* __restrict__ b1, const float* __restrict__ b2,
    const float* __restrict__ W3, const float* __restrict__ b3,
    float* __restrict__ out, int E)
{
  extern __shared__ char smem[];
  __bf16* stw = reinterpret_cast<__bf16*>(smem);
  __bf16* w1L = reinterpret_cast<__bf16*>(smem + W1_LDS_OFF);
  __bf16* w2L = reinterpret_cast<__bf16*>(smem + W2_LDS_OFF);

  const int tid  = threadIdx.x;
  const int wave = tid >> 5;
  const int lane = tid & 31;
  const int l16  = lane & 15;
  const int hi   = lane >> 4;                  // 0: lanes 0-15, 1: lanes 16-31
  __bf16* st = stw + wave * (16 * ASTRIDE);

  // ---- stage weights into LDS once per block (TDM with row padding) ---------
#if TDM_MODE
  if (wave == 0) {                             // wave 0 -> W1 (64 dw rows + 4 pad)
    tdm_load_2d(tdm_group0(W1_LDS_OFF, w1t_u),
                tdm_group1(64u, 256u, 5u /*64 dw*/, 3u /*4 dw*/));
#if __has_builtin(__builtin_amdgcn_s_wait_tensorcnt)
    __builtin_amdgcn_s_wait_tensorcnt(0);
#endif
  } else if (wave == 1) {                      // wave 1 -> W2 (128 dw rows + 4 pad)
    tdm_load_2d(tdm_group0(W2_LDS_OFF, w2t_u),
                tdm_group1(128u, 256u, 6u /*128 dw*/, 3u /*4 dw*/));
#if __has_builtin(__builtin_amdgcn_s_wait_tensorcnt)
    __builtin_amdgcn_s_wait_tensorcnt(0);
#endif
  }
#else
  {
    const uint4* g1 = reinterpret_cast<const uint4*>(w1t_u);   // 4096 x 16B
    const uint4* g2 = reinterpret_cast<const uint4*>(w2t_u);   // 16384 x 16B
    #pragma unroll 4
    for (int c = tid; c < 4096; c += 256) {
      const int row = c >> 4, off = c & 15;
      *reinterpret_cast<uint4*>(w1L + row * W1STRIDE + off * 8) = g1[c];
    }
    #pragma unroll 4
    for (int c = tid; c < 16384; c += 256) {
      const int row = c >> 5, off = c & 31;
      *reinterpret_cast<uint4*>(w2L + row * W2STRIDE + off * 8) = g2[c];
    }
  }
#endif
  __syncthreads();

  bf16x16 a2[2][8];                            // layer-2 A fragments, both row tiles

  #pragma unroll
  for (int m = 0; m < 2; ++m) {
    const int edgeBase = blockIdx.x * 256 + wave * 32 + m * 16;

    // ---- gather + u_mul_v -> bf16 staging [16 rows][128 cols] ---------------
    int eidx = edgeBase + l16;
    if (eidx >= E) eidx = E - 1;               // clamp tail (recompute duplicates)
    const int myIdx = hi ? dst[eidx] : src[eidx];

    #pragma unroll
    for (int e = 0; e < 16; ++e) {
      const int s = __shfl(myIdx, e);
      const int d = __shfl(myIdx, 16 + e);
      const float4 hs = *reinterpret_cast<const float4*>(h + (size_t)s * 128 + lane * 4);
      const float4 hd = *reinterpret_cast<const float4*>(h + (size_t)d * 128 + lane * 4);
      bf16x4 p;
      p[0] = (__bf16)(hs.x * hd.x);
      p[1] = (__bf16)(hs.y * hd.y);
      p[2] = (__bf16)(hs.z * hd.z);
      p[3] = (__bf16)(hs.w * hd.w);
      *reinterpret_cast<bf16x4*>(st + e * ASTRIDE + lane * 4) = p;
    }

    // ---- layer-1 A fragments (contiguous k order, same as B) --------------------------
    // lanes 0-15: K=[0..15]; lanes 16-31: K=[16..31]
    bf16x16 a1[4];
    #pragma unroll
    for (int kk = 0; kk < 4; ++kk) {
      const bf16x8 lo = *reinterpret_cast<const bf16x8*>(st + l16 * ASTRIDE + kk * 32 + hi * 16);
      const bf16x8 up = *reinterpret_cast<const bf16x8*>(st + l16 * ASTRIDE + kk * 32 + 8 + hi * 16);
      a1[kk] = __builtin_shufflevector(lo, up, 0, 1, 2, 3, 4, 5, 6, 7, 8, 9, 10, 11, 12, 13, 14, 15);
    }

    // ---- layer 1: [16x128] x [128x256] + b1, ReLU -> staging [16][256] ------
    // unroll 2 -> two independent WMMA accumulation chains in flight
    #pragma unroll 2
    for (int j = 0; j < 16; ++j) {
      const float bb = b1[j * 16 + l16];
      v8f acc;
      #pragma unroll
      for (int r = 0; r < 8; ++r) acc[r] = bb;
      #pragma unroll
      for (int kk = 0; kk < 4; ++kk) {
        const bf16x16 bf = *reinterpret_cast<const bf16x16*>(
            w1L + (j * 16 + l16) * W1STRIDE + kk * 32 + hi * 16);
        acc = __builtin_amdgcn_wmma_f32_16x16x32_bf16(
            false, a1[kk], false, bf, (short)0, acc, false, false);
      }
      #pragma unroll
      for (int r = 0; r < 8; ++r) {            // ReLU (v_med3) + bf16 scatter (C layout)
        st[(r + hi * 8) * ASTRIDE + j * 16 + l16] = (__bf16)relu1(acc[r]);
      }
    }

    // ---- layer-2 A fragments for this tile (K = 256 -> 8 k-steps) -----------
    #pragma unroll
    for (int kk = 0; kk < 8; ++kk) {
      const bf16x8 lo = *reinterpret_cast<const bf16x8*>(st + l16 * ASTRIDE + kk * 32 + hi * 16);
      const bf16x8 up = *reinterpret_cast<const bf16x8*>(st + l16 * ASTRIDE + kk * 32 + 8 + hi * 16);
      a2[m][kk] = __builtin_shufflevector(lo, up, 0, 1, 2, 3, 4, 5, 6, 7, 8, 9, 10, 11, 12, 13, 14, 15);
    }
  }

  // ---- layer 2 (+ReLU) with B reused across both row tiles; fused W3 dot ----
  float part[2][8];
  #pragma unroll
  for (int r = 0; r < 8; ++r) { part[0][r] = 0.0f; part[1][r] = 0.0f; }

  for (int j = 0; j < 16; ++j) {
    const float bb = b2[j * 16 + l16];
    v8f acc0, acc1;
    #pragma unroll
    for (int r = 0; r < 8; ++r) { acc0[r] = bb; acc1[r] = bb; }
    #pragma unroll
    for (int kk = 0; kk < 8; ++kk) {
      const bf16x16 bf = *reinterpret_cast<const bf16x16*>(
          w2L + (j * 16 + l16) * W2STRIDE + kk * 32 + hi * 16);
      acc0 = __builtin_amdgcn_wmma_f32_16x16x32_bf16(
          false, a2[0][kk], false, bf, (short)0, acc0, false, false);
      acc1 = __builtin_amdgcn_wmma_f32_16x16x32_bf16(
          false, a2[1][kk], false, bf, (short)0, acc1, false, false);
    }
    const float w3v = W3[j * 16 + l16];
    #pragma unroll
    for (int r = 0; r < 8; ++r) {
      part[0][r] += relu1(acc0[r]) * w3v;
      part[1][r] += relu1(acc1[r]) * w3v;
    }
  }

  // ---- reduce the column-partial dots across each 16-lane half --------------
  #pragma unroll
  for (int mm = 8; mm >= 1; mm >>= 1) {
    #pragma unroll
    for (int r = 0; r < 8; ++r) {
      part[0][r] += __shfl_xor(part[0][r], mm);
      part[1][r] += __shfl_xor(part[1][r], mm);
    }
  }

  const float b3v = b3[0];
  if (l16 == 0) {                              // lane 0 -> rows 0-7, lane 16 -> rows 8-15
    #pragma unroll
    for (int m = 0; m < 2; ++m) {
      #pragma unroll
      for (int r = 0; r < 8; ++r) {
        const int row = blockIdx.x * 256 + wave * 32 + m * 16 + hi * 8 + r;
        if (row < E) out[row] = part[m][r] + b3v;
      }
    }
  }
}

// ---------------------------------------------------------------------------
extern "C" void kernel_launch(void* const* d_in, const int* in_sizes, int n_in,
                              void* d_out, int out_size, void* d_ws, size_t ws_size,
                              hipStream_t stream) {
  const float* h   = (const float*)d_in[0];
  const int*   src = (const int*)d_in[1];
  const int*   dst = (const int*)d_in[2];
  const float* W1  = (const float*)d_in[3];
  const float* b1  = (const float*)d_in[4];
  const float* W2  = (const float*)d_in[5];
  const float* b2  = (const float*)d_in[6];
  const float* W3  = (const float*)d_in[7];
  const float* b3  = (const float*)d_in[8];
  float* out = (float*)d_out;
  const int E = in_sizes[1];

  unsigned short* w1t = (unsigned short*)d_ws;             // 256*128 bf16 = 64 KB
  unsigned short* w2t = w1t + 256 * 128;                   // 256*256 bf16 = 128 KB

  (void)hipFuncSetAttribute(reinterpret_cast<const void*>(edge_mlp),
                            hipFuncAttributeMaxDynamicSharedMemorySize, LDS_TOTAL);

  {
    const int total = 256 * 128 + 256 * 256;
    prep_weights<<<(total + 127) / 128, 128, 0, stream>>>(W1, W2, w1t, w2t);
  }
  {
    const int blocks = (E + 255) / 256;                    // 256 edges / block (8 waves x 32)
    edge_mlp<<<blocks, 256, LDS_TOTAL, stream>>>(h, src, dst, w1t, w2t, b1, b2, W3, b3, out, E);
  }
}
